// MCTS_GAT_36953898615235
// MI455X (gfx1250) — matrix operand order repaired and checked by reading, not compiled, not run on hardware
//
#include <hip/hip_runtime.h>
#include <hip/hip_bf16.h>

typedef __attribute__((ext_vector_type(16))) _Float16 v16h;
typedef __attribute__((ext_vector_type(8)))  _Float16 v8h;
typedef __attribute__((ext_vector_type(4)))  _Float16 v4h;
typedef __attribute__((ext_vector_type(8)))  float    v8f;

#define N_NODES 40000
#define N_EDGES 640000
#define N_MOVES 100000
#define HID 128

// WMMA f16 fragment layouts (CDNA5 ISA §7.12.2):
//  A 16x32: lane r=lane&15 -> M row; half=lane>>4:
//    slots 0-7  = K (k0 + 8*half .. +7), slots 8-15 = K (k0+16+8*half .. +7)
//  B 32x16 (from K-major transposed LDS): lane r -> N col; half:
//    slots 0-7  = K (k0 + 8*half .. +7), slots 8-15 = K (k0+16+8*half .. +7)
union FragU { v16h v; v8h h[2]; };

__device__ __forceinline__ void atomicMaxF(float* addr, float val) {
    if (val >= 0.0f)
        atomicMax((int*)addr, __float_as_int(val));
    else
        atomicMin((unsigned int*)addr, __float_as_uint(val));
}

__device__ __forceinline__ v4h cvt4(float a, float b, float c, float d) {
    v4h r = {(_Float16)a, (_Float16)b, (_Float16)c, (_Float16)d};
    return r;
}

// ------------------------------------------------------------------
// C[M,128] = A[M,K] @ B[K,128], M multiple of 16.  One 16-row tile per
// block; 8 waves each own one 16-col N tile.  A row-major, B transposed
// in LDS (f16); staging is float4-global + v4h-LDS vectorized.
// ------------------------------------------------------------------
template <int K, int KP>
__global__ void gemm_tile_wmma(const float* __restrict__ A, const float* __restrict__ B,
                               float* __restrict__ C) {
    __shared__ _Float16 sA[16][KP];
    __shared__ _Float16 sBt[128][KP];     // transposed: [n][k]
    const int tid = threadIdx.x;
    const int m0  = blockIdx.x * 16;

    if constexpr (K % 4 != 0) {
        // small-K path (K=11): zero-pad then elementwise staging
        for (int i = tid; i < 16 * KP; i += 256)  sA[i / KP][i % KP]  = (_Float16)0.0f;
        for (int i = tid; i < 128 * KP; i += 256) sBt[i / KP][i % KP] = (_Float16)0.0f;
        __syncthreads();
        for (int i = tid; i < 16 * K; i += 256) {
            const int row = i / K, col = i % K;
            sA[row][col] = (_Float16)A[(size_t)(m0 + row) * K + col];
        }
        for (int i = tid; i < K * 128; i += 256) {
            const int k = i >> 7, n = i & 127;
            sBt[n][k] = (_Float16)B[(size_t)k * 128 + n];
        }
    } else {
        // A: float4 loads, v4h stores (coalesced both sides)
        for (int i = tid; i < 16 * (K / 4); i += 256) {
            const int row = i / (K / 4), c4 = i % (K / 4);
            const float4 v = *(const float4*)&A[(size_t)(m0 + row) * K + c4 * 4];
            *(v4h*)&sA[row][c4 * 4] = cvt4(v.x, v.y, v.z, v.w);
        }
        // B: 4x4 tile transpose per thread
        for (int t = tid; t < (K / 4) * 32; t += 256) {
            const int kt = t >> 5, nt = t & 31;
            const int k = kt * 4, n = nt * 4;
            const float4 r0 = *(const float4*)&B[(size_t)(k + 0) * 128 + n];
            const float4 r1 = *(const float4*)&B[(size_t)(k + 1) * 128 + n];
            const float4 r2 = *(const float4*)&B[(size_t)(k + 2) * 128 + n];
            const float4 r3 = *(const float4*)&B[(size_t)(k + 3) * 128 + n];
            *(v4h*)&sBt[n + 0][k] = cvt4(r0.x, r1.x, r2.x, r3.x);
            *(v4h*)&sBt[n + 1][k] = cvt4(r0.y, r1.y, r2.y, r3.y);
            *(v4h*)&sBt[n + 2][k] = cvt4(r0.z, r1.z, r2.z, r3.z);
            *(v4h*)&sBt[n + 3][k] = cvt4(r0.w, r1.w, r2.w, r3.w);
        }
    }
    __syncthreads();

    const int wave = tid >> 5, lane = tid & 31;
    const int half = lane >> 4, r = lane & 15;
    const int n0   = wave * 16;

    v8f c = {};
#pragma unroll
    for (int k0 = 0; k0 < KP; k0 += 32) {
        FragU a, b;
        a.h[0] = *(const v8h*)(&sA[r][k0 + 8 * half]);
        a.h[1] = *(const v8h*)(&sA[r][k0 + 16 + 8 * half]);
        b.h[0] = *(const v8h*)(&sBt[n0 + r][k0 + 8 * half]);
        b.h[1] = *(const v8h*)(&sBt[n0 + r][k0 + 16 + 8 * half]);
        c = __builtin_amdgcn_wmma_f32_16x16x32_f16(false, a.v, false, b.v,
                                                   (short)0, c, false, false);
    }
#pragma unroll
    for (int rr = 0; rr < 8; ++rr)
        C[(size_t)(m0 + rr + 8 * half) * 128 + n0 + r] = c[rr];
}

// ---- per-node attention halves + per-node softmax-state init ----
__global__ void attn_coef(const float* __restrict__ h,
                          const float* __restrict__ asrc, const float* __restrict__ adst,
                          float* __restrict__ esrc, float* __restrict__ edst,
                          float* __restrict__ mmax, float* __restrict__ ssum, int n) {
    const int w = (blockIdx.x * blockDim.x + threadIdx.x) >> 5;
    const int lane = threadIdx.x & 31;
    if (w >= n) return;
    const float4 hv = ((const float4*)(h + (size_t)w * HID))[lane];
    const float4 sa = ((const float4*)asrc)[lane];
    const float4 da = ((const float4*)adst)[lane];
    float ps = hv.x * sa.x + hv.y * sa.y + hv.z * sa.z + hv.w * sa.w;
    float pd = hv.x * da.x + hv.y * da.y + hv.z * da.z + hv.w * da.w;
    for (int off = 16; off; off >>= 1) {
        ps += __shfl_down(ps, off, 32);
        pd += __shfl_down(pd, off, 32);
    }
    if (lane == 0) {
        esrc[w] = ps;
        edst[w] = pd;
        mmax[w] = -3.402823466e38f;
        ssum[w] = 0.0f;
    }
}

__global__ void zero_f4(float4* __restrict__ p, int n4) {
    for (int i = blockIdx.x * blockDim.x + threadIdx.x; i < n4;
         i += gridDim.x * blockDim.x)
        p[i] = make_float4(0.f, 0.f, 0.f, 0.f);
}

__global__ void edge_score(const int* __restrict__ src, const int* __restrict__ dst,
                           const float* __restrict__ esrc, const float* __restrict__ edst,
                           float* __restrict__ e, float* __restrict__ mmax, int E) {
    const int i = blockIdx.x * blockDim.x + threadIdx.x;
    if (i >= E) return;
    float v = esrc[src[i]] + edst[dst[i]];
    v = (v > 0.0f) ? v : 0.2f * v;          // leaky_relu 0.2
    e[i] = v;
    atomicMaxF(&mmax[dst[i]], v);
}

__global__ void edge_exp(const int* __restrict__ dst, float* __restrict__ e,
                         const float* __restrict__ mmax, float* __restrict__ ssum, int E) {
    const int i = blockIdx.x * blockDim.x + threadIdx.x;
    if (i >= E) return;
    const int d = dst[i];
    const float ex = __expf(e[i] - mmax[d]);
    e[i] = ex;
    atomicAdd(&ssum[d], ex);
}

__global__ void edge_agg(const int* __restrict__ src, const int* __restrict__ dst,
                         const float* __restrict__ e, const float* __restrict__ ssum,
                         const float* __restrict__ h, float* __restrict__ agg, int E) {
    const int w = (blockIdx.x * blockDim.x + threadIdx.x) >> 5;
    const int lane = threadIdx.x & 31;
    if (w >= E) return;
    const int s = src[w], d = dst[w];
    const float alpha = e[w] / (ssum[d] + 1e-16f);
    const float4 v = ((const float4*)(h + (size_t)s * HID))[lane];
    float* out = agg + (size_t)d * HID + lane * 4;
    atomicAdd(out + 0, alpha * v.x);
    atomicAdd(out + 1, alpha * v.y);
    atomicAdd(out + 2, alpha * v.z);
    atomicAdd(out + 3, alpha * v.w);
}

__global__ void bias_relu4(const float4* __restrict__ agg, const float4* __restrict__ b,
                           float4* __restrict__ out, int n4) {
    const int i = blockIdx.x * blockDim.x + threadIdx.x;
    if (i >= n4) return;
    const float4 a = agg[i];
    const float4 bb = b[i & 31];           // 128 ch = 32 float4
    out[i] = make_float4(fmaxf(a.x + bb.x, 0.f), fmaxf(a.y + bb.y, 0.f),
                         fmaxf(a.z + bb.z, 0.f), fmaxf(a.w + bb.w, 0.f));
}

// ------------------------------------------------------------------
// Fused policy head: 32 moves/block (3125 blocks).
// gather -> As[32][256] f16; per K-chunk stage P1 transposed in LDS;
// 2 M-tiles x 8 N-tiles of WMMA; bias+ReLU -> Hid; 8-lane-per-row P2 dot.
// ------------------------------------------------------------------
__global__ void policy_head(const float* __restrict__ h,
                            const int* __restrict__ ms, const int* __restrict__ md,
                            const float* __restrict__ P1, const float* __restrict__ pb1,
                            const float* __restrict__ P2, const float* __restrict__ pb2,
                            float* __restrict__ logits) {
    __shared__ _Float16 As[32][256];
    __shared__ _Float16 sPt[128][32];     // P1 chunk transposed: [n][k]
    __shared__ float    Hid[32][HID];
    __shared__ float    red[32][8];
    const int tid   = threadIdx.x;
    const int move0 = blockIdx.x * 32;

    // gather 32 move rows (h[src] || h[dst]) as float4 -> v4h
    for (int i = tid; i < 32 * 64; i += 256) {
        const int row = i >> 6, q = i & 63;
        const int mv   = move0 + row;
        const int node = (q < 32) ? ms[mv] : md[mv];
        const float4 v = ((const float4*)(h + (size_t)node * HID))[q & 31];
        *(v4h*)&As[row][q * 4] = cvt4(v.x, v.y, v.z, v.w);
    }

    const int wave = tid >> 5, lane = tid & 31;
    const int half = lane >> 4, r = lane & 15;
    const int n0   = wave * 16;

    v8f c0 = {}, c1 = {};
    for (int k0 = 0; k0 < 256; k0 += 32) {
        __syncthreads();                  // also orders As writes before first use
        // stage P1[k0..k0+31][0..127] transposed, 4x4 tiles (256 tiles, 1/thread)
        {
            const int kt = tid >> 5, nt = tid & 31;
            const int k = k0 + kt * 4, n = nt * 4;
            const float4 r0 = *(const float4*)&P1[(size_t)(k + 0) * HID + n];
            const float4 r1 = *(const float4*)&P1[(size_t)(k + 1) * HID + n];
            const float4 r2 = *(const float4*)&P1[(size_t)(k + 2) * HID + n];
            const float4 r3 = *(const float4*)&P1[(size_t)(k + 3) * HID + n];
            const int kk = kt * 4;
            *(v4h*)&sPt[n + 0][kk] = cvt4(r0.x, r1.x, r2.x, r3.x);
            *(v4h*)&sPt[n + 1][kk] = cvt4(r0.y, r1.y, r2.y, r3.y);
            *(v4h*)&sPt[n + 2][kk] = cvt4(r0.z, r1.z, r2.z, r3.z);
            *(v4h*)&sPt[n + 3][kk] = cvt4(r0.w, r1.w, r2.w, r3.w);
        }
        __syncthreads();
        FragU a0, a1, b;
        b.h[0]  = *(const v8h*)(&sPt[n0 + r][8 * half]);
        b.h[1]  = *(const v8h*)(&sPt[n0 + r][16 + 8 * half]);
        a0.h[0] = *(const v8h*)(&As[r][k0 + 8 * half]);
        a0.h[1] = *(const v8h*)(&As[r][k0 + 16 + 8 * half]);
        a1.h[0] = *(const v8h*)(&As[16 + r][k0 + 8 * half]);
        a1.h[1] = *(const v8h*)(&As[16 + r][k0 + 16 + 8 * half]);
        c0 = __builtin_amdgcn_wmma_f32_16x16x32_f16(false, a0.v, false, b.v,
                                                    (short)0, c0, false, false);
        c1 = __builtin_amdgcn_wmma_f32_16x16x32_f16(false, a1.v, false, b.v,
                                                    (short)0, c1, false, false);
    }

    const float bval = pb1[n0 + r];
#pragma unroll
    for (int rr = 0; rr < 8; ++rr) {
        Hid[rr + 8 * half][n0 + r]      = fmaxf(c0[rr] + bval, 0.0f);
        Hid[16 + rr + 8 * half][n0 + r] = fmaxf(c1[rr] + bval, 0.0f);
    }
    __syncthreads();

    // logits: 8 lanes per move row
    {
        const int row = tid >> 3, l8 = tid & 7;
        float p = 0.0f;
        for (int n = l8; n < HID; n += 8) p += Hid[row][n] * P2[n];
        red[row][l8] = p;
    }
    __syncthreads();
    if (tid < 32) {
        float acc = pb2[0];
#pragma unroll
        for (int j = 0; j < 8; ++j) acc += red[tid][j];
        logits[move0 + tid] = acc;
    }
}

// ---- softmax helpers ----
__global__ void init_head(float* __restrict__ scal, float* __restrict__ gvec) {
    const int t = threadIdx.x;
    if (t == 0) { scal[0] = -3.402823466e38f; scal[1] = 0.0f; }
    if (t < HID) gvec[t] = 0.0f;
}

__global__ void red_max(const float* __restrict__ v, float* __restrict__ gmax, int n) {
    __shared__ float s[256];
    float m = -3.402823466e38f;
    for (int i = blockIdx.x * blockDim.x + threadIdx.x; i < n;
         i += gridDim.x * blockDim.x)
        m = fmaxf(m, v[i]);
    s[threadIdx.x] = m; __syncthreads();
    for (int o = 128; o; o >>= 1) {
        if (threadIdx.x < o) s[threadIdx.x] = fmaxf(s[threadIdx.x], s[threadIdx.x + o]);
        __syncthreads();
    }
    if (threadIdx.x == 0) atomicMaxF(gmax, s[0]);
}

__global__ void red_sumexp(const float* __restrict__ v, float* __restrict__ scal, int n) {
    __shared__ float s[256];
    const float gm = scal[0];
    float a = 0.0f;
    for (int i = blockIdx.x * blockDim.x + threadIdx.x; i < n;
         i += gridDim.x * blockDim.x)
        a += __expf(v[i] - gm);
    s[threadIdx.x] = a; __syncthreads();
    for (int o = 128; o; o >>= 1) {
        if (threadIdx.x < o) s[threadIdx.x] += s[threadIdx.x + o];
        __syncthreads();
    }
    if (threadIdx.x == 0) atomicAdd(&scal[1], s[0]);
}

__global__ void write_probs(const float* __restrict__ logits, const float* __restrict__ scal,
                            float* __restrict__ out, int n) {
    const int i = blockIdx.x * blockDim.x + threadIdx.x;
    if (i < n) out[i] = __expf(logits[i] - scal[0]) / scal[1];
}

// ---- value head ----
__global__ void pool_sum(const float* __restrict__ h, float* __restrict__ g, int n) {
    const int c = threadIdx.x;   // 128
    float acc = 0.0f;
    for (int nd = blockIdx.x; nd < n; nd += gridDim.x)
        acc += h[(size_t)nd * HID + c];
    atomicAdd(&g[c], acc);
}

__global__ void value_head(const float* __restrict__ g,
                           const float* __restrict__ V1, const float* __restrict__ vb1,
                           const float* __restrict__ V2, const float* __restrict__ vb2,
                           float* __restrict__ out, float invN) {
    __shared__ float sg[HID];
    __shared__ float sr[HID];
    const int c = threadIdx.x;
    sg[c] = g[c] * invN;
    __syncthreads();
    float acc = vb1[c];
    for (int k = 0; k < HID; ++k) acc += sg[k] * V1[(size_t)k * HID + c];
    acc = fmaxf(acc, 0.0f);
    sr[c] = acc * V2[c];
    __syncthreads();
    for (int o = 64; o; o >>= 1) {
        if (c < o) sr[c] += sr[c + o];
        __syncthreads();
    }
    if (c == 0) out[0] = tanhf(sr[0] + vb2[0]);
}

extern "C" void kernel_launch(void* const* d_in, const int* in_sizes, int n_in,
                              void* d_out, int out_size, void* d_ws, size_t ws_size,
                              hipStream_t stream) {
    (void)in_sizes; (void)n_in; (void)out_size; (void)ws_size;
    const float* x   = (const float*)d_in[0];
    const int*   ei  = (const int*)d_in[1];
    const int*   lm  = (const int*)d_in[2];
    // d_in[3] = batch (all zeros, single graph) -- unused
    const float* W1  = (const float*)d_in[4];
    const float* a1s = (const float*)d_in[5];
    const float* a1d = (const float*)d_in[6];
    const float* b1  = (const float*)d_in[7];
    const float* W2  = (const float*)d_in[8];
    const float* a2s = (const float*)d_in[9];
    const float* a2d = (const float*)d_in[10];
    const float* b2  = (const float*)d_in[11];
    const float* P1  = (const float*)d_in[12];
    const float* pb1 = (const float*)d_in[13];
    const float* P2  = (const float*)d_in[14];
    const float* pb2 = (const float*)d_in[15];
    const float* V1  = (const float*)d_in[16];
    const float* vb1 = (const float*)d_in[17];
    const float* V2  = (const float*)d_in[18];
    const float* vb2 = (const float*)d_in[19];
    float* out = (float*)d_out;

    const int* src = ei;
    const int* dst = ei + N_EDGES;
    const int* ms  = lm;
    const int* md  = lm + N_MOVES;

    // workspace layout (floats)
    float* ws = (float*)d_ws;
    const size_t NH = (size_t)N_NODES * HID;
    float* bufA   = ws;                 // h_pre (attention/aggregation input)
    float* bufB   = bufA + NH;          // activated h
    float* bufC   = bufB + NH;          // aggregation accumulator
    float* eBuf   = bufC + NH;          // per-edge score / exp
    float* esrc   = eBuf + N_EDGES;
    float* edst   = esrc + N_NODES;
    float* mmax   = edst + N_NODES;
    float* ssum   = mmax + N_NODES;
    float* logits = ssum + N_NODES;
    float* scal   = logits + N_MOVES;   // [gmax, gsum]
    float* gvec   = scal + 8;           // pooled vector (128)

    const dim3 blk(256);
    const int  gTiles = N_NODES / 16;               // 2500, exact
    const int  gAttn  = (N_NODES * 32) / 256;       // wave per node
    const int  gEdgeT = (N_EDGES + 255) / 256;      // thread per edge
    const int  gEdgeW = (N_EDGES * 32) / 256;       // wave per edge
    const int  gNH4   = (int)(NH / 4 / 256);        // float4 elementwise

    // -------- layer 1 --------
    gemm_tile_wmma<11, 32><<<gTiles, blk, 0, stream>>>(x, W1, bufA);
    attn_coef<<<gAttn, blk, 0, stream>>>(bufA, a1s, a1d, esrc, edst, mmax, ssum, N_NODES);
    zero_f4<<<2048, blk, 0, stream>>>((float4*)bufC, (int)(NH / 4));
    edge_score<<<gEdgeT, blk, 0, stream>>>(src, dst, esrc, edst, eBuf, mmax, N_EDGES);
    edge_exp<<<gEdgeT, blk, 0, stream>>>(dst, eBuf, mmax, ssum, N_EDGES);
    edge_agg<<<gEdgeW, blk, 0, stream>>>(src, dst, eBuf, ssum, bufA, bufC, N_EDGES);
    bias_relu4<<<gNH4, blk, 0, stream>>>((const float4*)bufC, (const float4*)b1,
                                         (float4*)bufB, (int)(NH / 4));

    // -------- layer 2 --------
    gemm_tile_wmma<128, 128><<<gTiles, blk, 0, stream>>>(bufB, W2, bufA);
    attn_coef<<<gAttn, blk, 0, stream>>>(bufA, a2s, a2d, esrc, edst, mmax, ssum, N_NODES);
    zero_f4<<<2048, blk, 0, stream>>>((float4*)bufC, (int)(NH / 4));
    edge_score<<<gEdgeT, blk, 0, stream>>>(src, dst, esrc, edst, eBuf, mmax, N_EDGES);
    edge_exp<<<gEdgeT, blk, 0, stream>>>(dst, eBuf, mmax, ssum, N_EDGES);
    edge_agg<<<gEdgeW, blk, 0, stream>>>(src, dst, eBuf, ssum, bufA, bufC, N_EDGES);
    bias_relu4<<<gNH4, blk, 0, stream>>>((const float4*)bufC, (const float4*)b2,
                                         (float4*)bufB, (int)(NH / 4));

    // -------- heads --------
    init_head<<<1, blk, 0, stream>>>(scal, gvec);
    policy_head<<<N_MOVES / 32, blk, 0, stream>>>(bufB, ms, md, P1, pb1, P2, pb2, logits);
    red_max<<<200, blk, 0, stream>>>(logits, &scal[0], N_MOVES);
    red_sumexp<<<200, blk, 0, stream>>>(logits, scal, N_MOVES);
    write_probs<<<(N_MOVES + 255) / 256, blk, 0, stream>>>(logits, scal, out + 1, N_MOVES);

    pool_sum<<<256, HID, 0, stream>>>(bufB, gvec, N_NODES);
    value_head<<<1, HID, 0, stream>>>(gvec, V1, vb1, V2, vb2, out, 1.0f / N_NODES);
}
